// LlamaAttention_36584531427989
// MI455X (gfx1250) — compile-verified, matrix operand order checked by reading, not run on hardware
//
#include <hip/hip_runtime.h>

// ---------------------------------------------------------------------------
// Types / constants
// ---------------------------------------------------------------------------
typedef __attribute__((ext_vector_type(16))) __bf16 v16bf;
typedef __attribute__((ext_vector_type(8)))  float  v8f;

#define S_LEN 2048
#define HID   4096
#define NQH   32
#define NKVH  8
#define HD    128

__device__ __forceinline__ unsigned short f2bf(float f) {
  unsigned u = __float_as_uint(f);
  u += 0x7FFFu + ((u >> 16) & 1u);          // round-to-nearest-even
  return (unsigned short)(u >> 16);
}

__device__ __forceinline__ v16bf ldfrag(const unsigned short* p) {
  return *(const v16bf*)__builtin_assume_aligned(p, 32);
}
__device__ __forceinline__ v16bf ldfrag_lds(const unsigned short* p) {
  return *(const v16bf*)__builtin_assume_aligned(p, 32);
}

__device__ __forceinline__ v8f wmma_bf16(v16bf a, v16bf b, v8f c) {
  // D = A(16x32 bf16) * B(32x16 bf16) + C(16x16 f32)
  return __builtin_amdgcn_wmma_f32_16x16x32_bf16(
      /*neg_a=*/false, a, /*neg_b=*/false, b,
      /*c_mod=*/(short)0, c, /*reuse_a=*/false, /*reuse_b=*/false);
}

__device__ __forceinline__ float red_max16(float v) {
  v = fmaxf(v, __shfl_xor(v, 1, 32));
  v = fmaxf(v, __shfl_xor(v, 2, 32));
  v = fmaxf(v, __shfl_xor(v, 4, 32));
  v = fmaxf(v, __shfl_xor(v, 8, 32));
  return v;
}
__device__ __forceinline__ float red_sum16(float v) {
  v += __shfl_xor(v, 1, 32);
  v += __shfl_xor(v, 2, 32);
  v += __shfl_xor(v, 4, 32);
  v += __shfl_xor(v, 8, 32);
  return v;
}

// ---------------------------------------------------------------------------
// f32 -> bf16 conversion of hidden states
// ---------------------------------------------------------------------------
__global__ __launch_bounds__(256) void k_f32_to_bf16(
    const float* __restrict__ x, unsigned short* __restrict__ y, int n) {
  int i = blockIdx.x * 256 + threadIdx.x;
  if (i < n) y[i] = f2bf(x[i]);
}

// ---------------------------------------------------------------------------
// Pack a KxN row-major f32 weight matrix into bf16 B-fragment layout.
// Tile (nt,kt) is 32x16, stored as 32 lanes x 16 contiguous bf16:
//   packed[((nt*Ktiles)+kt)*512 + lane*16 + j]
//       = W[kt*32 + (lane>>4)*16 + j][nt*16 + (lane&15)]
// ---------------------------------------------------------------------------
__global__ __launch_bounds__(256) void k_pack_w(
    const float* __restrict__ w, unsigned short* __restrict__ p, int K, int N) {
  int idx = blockIdx.x * 256 + threadIdx.x;
  if (idx >= K * N) return;
  int Ktiles = K >> 5;
  int j    = idx & 15;
  int lane = (idx >> 4) & 31;
  int t    = idx >> 9;
  int kt   = t % Ktiles;
  int nt   = t / Ktiles;
  int k = kt * 32 + ((lane >> 4) << 4) + j;
  int n = (nt << 4) + (lane & 15);
  p[idx] = f2bf(w[(size_t)k * N + n]);
}

// ---------------------------------------------------------------------------
// Fused QKV projection + RoPE.  One wave per (head-slot, 32-query block).
//   hs in [0,32)      : Q head  -> rope -> Qr[u][s][d]
//   hs in [32,40)     : K head  -> rope -> Kr[v][s][d]
//   hs in [40,48)     : V head  ->        Vt[v][d][s]  (transposed for PV)
// ---------------------------------------------------------------------------
__global__ __launch_bounds__(32) void k_qkv_rope(
    const unsigned short* __restrict__ Xb,
    const unsigned short* __restrict__ WqP,
    const unsigned short* __restrict__ WkP,
    const unsigned short* __restrict__ WvP,
    unsigned short* __restrict__ Qr,
    unsigned short* __restrict__ Kr,
    unsigned short* __restrict__ Vt) {
  const int lane = threadIdx.x;
  const int g16 = lane >> 4, l16 = lane & 15;
  const int qb = blockIdx.x;   // 32-query block
  const int hs = blockIdx.y;   // head slot 0..47

  const unsigned short* Wp;
  int ntBase, isRope, kvh = 0;
  if (hs < NQH)              { Wp = WqP; ntBase = hs * 8;              isRope = 1; }
  else if (hs < NQH + NKVH)  { Wp = WkP; kvh = hs - NQH;        ntBase = kvh * 8; isRope = 1; }
  else                       { Wp = WvP; kvh = hs - NQH - NKVH; ntBase = kvh * 8; isRope = 0; }

  v8f acc[2][8] = {};
  const int Ktiles = HID / 32;
  const unsigned short* a0p = Xb + (size_t)(qb * 32 + l16)      * HID + g16 * 16;
  const unsigned short* a1p = Xb + (size_t)(qb * 32 + 16 + l16) * HID + g16 * 16;
  const unsigned short* bp  = Wp + (size_t)ntBase * Ktiles * 512 + lane * 16;

  for (int kt = 0; kt < Ktiles; ++kt) {
    v16bf a0 = ldfrag(a0p + kt * 32);
    v16bf a1 = ldfrag(a1p + kt * 32);
    // streaming prefetch 16 k-tiles ahead (global_prefetch_b8)
    if (kt + 16 < Ktiles) {
      __builtin_prefetch(a0p + (kt + 16) * 32, 0, 1);
      __builtin_prefetch(bp + (size_t)(kt + 16) * 512, 0, 1);
    }
#pragma unroll
    for (int nt = 0; nt < 8; ++nt) {
      v16bf b = ldfrag(bp + ((size_t)nt * Ktiles + kt) * 512);
      acc[0][nt] = wmma_bf16(a0, b, acc[0][nt]);
      acc[1][nt] = wmma_bf16(a1, b, acc[1][nt]);
    }
  }

  if (isRope) {
    unsigned short* Out = (hs < NQH) ? (Qr + (size_t)hs * S_LEN * HD)
                                     : (Kr + (size_t)kvh * S_LEN * HD);
#pragma unroll
    for (int mt = 0; mt < 2; ++mt)
#pragma unroll
      for (int r = 0; r < 8; ++r) {
        int srow = qb * 32 + mt * 16 + r + g16 * 8;
        float pos = (float)srow;
#pragma unroll
        for (int ntl = 0; ntl < 4; ++ntl) {
          int h = ntl * 16 + l16;                       // 0..63
          float ang = pos * __powf(10000.0f, -(float)h * (1.0f / 64.0f));
          float sn, cs;
          __sincosf(ang, &sn, &cs);
          float x1 = acc[mt][ntl][r];
          float x2 = acc[mt][ntl + 4][r];
          Out[(size_t)srow * HD + h]      = f2bf(x1 * cs - x2 * sn);
          Out[(size_t)srow * HD + h + 64] = f2bf(x2 * cs + x1 * sn);
        }
      }
  } else {
    unsigned short* Out = Vt + (size_t)kvh * HD * S_LEN;
#pragma unroll
    for (int mt = 0; mt < 2; ++mt)
#pragma unroll
      for (int r = 0; r < 8; ++r) {
        int srow = qb * 32 + mt * 16 + r + g16 * 8;
#pragma unroll
        for (int nt = 0; nt < 8; ++nt) {
          int d = nt * 16 + l16;
          Out[(size_t)d * S_LEN + srow] = f2bf(acc[mt][nt][r]);
        }
      }
  }
}

// ---------------------------------------------------------------------------
// Causal flash attention. One wave per (q-head, 32-query block).
// Scores: A = Q rows (register-resident), B = K rows (row-major == K^T frags).
// PV:     A = P (via LDS round-trip),  B = Vt columns (contiguous).
// ---------------------------------------------------------------------------
#define LDS_STRIDE 48  // ushorts per row; 96B keeps 32B-aligned b128 LDS loads

__global__ __launch_bounds__(32) void k_attn(
    const unsigned short* __restrict__ Qr,
    const unsigned short* __restrict__ Kr,
    const unsigned short* __restrict__ Vt,
    unsigned short* __restrict__ AO) {
  __shared__ unsigned short ldsP[32 * LDS_STRIDE];

  const int lane = threadIdx.x;
  const int g16 = lane >> 4, l16 = lane & 15;
  const int qb = blockIdx.x;       // 32-query block
  const int u  = blockIdx.y;       // q head
  const int kvh = u >> 2;          // GQA: 4 q heads per kv head

  const unsigned short* Qh = Qr + (size_t)u   * S_LEN * HD;
  const unsigned short* Kh = Kr + (size_t)kvh * S_LEN * HD;
  const unsigned short* Vh = Vt + (size_t)kvh * HD * S_LEN;

  const float scale = 0.08838834764831845f;   // 1/sqrt(128)

  // Q fragments resident across the whole kb loop (64 VGPRs)
  v16bf qf[2][4];
#pragma unroll
  for (int mt = 0; mt < 2; ++mt)
#pragma unroll
    for (int kt = 0; kt < 4; ++kt)
      qf[mt][kt] =
          ldfrag(Qh + (size_t)(qb * 32 + mt * 16 + l16) * HD + kt * 32 + g16 * 16);

  v8f acc[2][8] = {};
  float mstate[2][8], lstate[2][8];
#pragma unroll
  for (int mt = 0; mt < 2; ++mt)
#pragma unroll
    for (int r = 0; r < 8; ++r) { mstate[mt][r] = -1e30f; lstate[mt][r] = 0.0f; }

  for (int kb = 0; kb <= qb; ++kb) {
    // ---- scores S = Q * K^T (2 q-subtiles x 2 key-subtiles) ----
    v8f sc[2][2] = {};
#pragma unroll
    for (int kt = 0; kt < 4; ++kt) {
      v16bf kf0 = ldfrag(Kh + (size_t)(kb * 32 + l16)      * HD + kt * 32 + g16 * 16);
      v16bf kf1 = ldfrag(Kh + (size_t)(kb * 32 + 16 + l16) * HD + kt * 32 + g16 * 16);
      sc[0][0] = wmma_bf16(qf[0][kt], kf0, sc[0][0]);
      sc[0][1] = wmma_bf16(qf[0][kt], kf1, sc[0][1]);
      sc[1][0] = wmma_bf16(qf[1][kt], kf0, sc[1][0]);
      sc[1][1] = wmma_bf16(qf[1][kt], kf1, sc[1][1]);
    }

    // ---- scale + causal mask (only diagonal block needs it) ----
    const int diag = (kb == qb);
#pragma unroll
    for (int mt = 0; mt < 2; ++mt)
#pragma unroll
      for (int ntk = 0; ntk < 2; ++ntk)
#pragma unroll
        for (int r = 0; r < 8; ++r) {
          float v = sc[mt][ntk][r] * scale;
          if (diag) {
            int key  = kb * 32 + ntk * 16 + l16;
            int qrow = qb * 32 + mt * 16 + r + g16 * 8;
            if (key > qrow) v = -1e30f;
          }
          sc[mt][ntk][r] = v;
        }

    // ---- online softmax update ----
    float alpha[2][8];
#pragma unroll
    for (int mt = 0; mt < 2; ++mt)
#pragma unroll
      for (int r = 0; r < 8; ++r) {
        float rmax = red_max16(fmaxf(sc[mt][0][r], sc[mt][1][r]));
        float mnew = fmaxf(mstate[mt][r], rmax);
        alpha[mt][r] = __expf(mstate[mt][r] - mnew);
        mstate[mt][r] = mnew;
      }
#pragma unroll
    for (int mt = 0; mt < 2; ++mt)
#pragma unroll
      for (int ntk = 0; ntk < 2; ++ntk)
#pragma unroll
        for (int r = 0; r < 8; ++r)
          sc[mt][ntk][r] = __expf(sc[mt][ntk][r] - mstate[mt][r]);
#pragma unroll
    for (int mt = 0; mt < 2; ++mt)
#pragma unroll
      for (int r = 0; r < 8; ++r) {
        float rs = red_sum16(sc[mt][0][r] + sc[mt][1][r]);
        lstate[mt][r] = lstate[mt][r] * alpha[mt][r] + rs;
      }
#pragma unroll
    for (int mt = 0; mt < 2; ++mt)
#pragma unroll
      for (int nt = 0; nt < 8; ++nt)
#pragma unroll
        for (int r = 0; r < 8; ++r)
          acc[mt][nt][r] *= alpha[mt][r];

    // ---- P -> LDS (C-layout -> A-layout re-stripe) ----
    asm volatile("s_wait_dscnt 0x0" ::: "memory");   // WAR vs previous kb's reads
#pragma unroll
    for (int mt = 0; mt < 2; ++mt)
#pragma unroll
      for (int ntk = 0; ntk < 2; ++ntk)
#pragma unroll
        for (int r = 0; r < 8; ++r)
          ldsP[(mt * 16 + r + g16 * 8) * LDS_STRIDE + ntk * 16 + l16] =
              f2bf(sc[mt][ntk][r]);
    asm volatile("s_wait_dscnt 0x0" ::: "memory");   // RAW before fragment reads

    v16bf p0 = ldfrag_lds(ldsP + (0 * 16 + l16) * LDS_STRIDE + g16 * 16);
    v16bf p1 = ldfrag_lds(ldsP + (1 * 16 + l16) * LDS_STRIDE + g16 * 16);

    // ---- O += P * V ----
#pragma unroll
    for (int nt = 0; nt < 8; ++nt) {
      v16bf vf = ldfrag(Vh + (size_t)(nt * 16 + l16) * S_LEN + kb * 32 + g16 * 16);
      acc[0][nt] = wmma_bf16(p0, vf, acc[0][nt]);
      acc[1][nt] = wmma_bf16(p1, vf, acc[1][nt]);
    }
  }

  // ---- normalize + store attention output (bf16, [s][u*128+d]) ----
#pragma unroll
  for (int mt = 0; mt < 2; ++mt)
#pragma unroll
    for (int r = 0; r < 8; ++r) {
      float inv = 1.0f / lstate[mt][r];
      int srow = qb * 32 + mt * 16 + r + g16 * 8;
#pragma unroll
      for (int nt = 0; nt < 8; ++nt)
        AO[(size_t)srow * HID + u * HD + nt * 16 + l16] = f2bf(acc[mt][nt][r] * inv);
    }
}

// ---------------------------------------------------------------------------
// Output projection: out[s][m] = sum_{u,h} attn[s][u*128+h] * wo[u*128+h][m]
// One wave per 32x64 output tile.
// ---------------------------------------------------------------------------
__global__ __launch_bounds__(32) void k_oproj(
    const unsigned short* __restrict__ AO,
    const unsigned short* __restrict__ WoP,
    float* __restrict__ Out) {
  const int lane = threadIdx.x;
  const int g16 = lane >> 4, l16 = lane & 15;
  const int nb = blockIdx.x;   // 64-col block
  const int mb = blockIdx.y;   // 32-row block
  const int Ktiles = HID / 32;

  const unsigned short* a0p = AO + (size_t)(mb * 32 + l16)      * HID + g16 * 16;
  const unsigned short* a1p = AO + (size_t)(mb * 32 + 16 + l16) * HID + g16 * 16;
  const unsigned short* bp  = WoP + (size_t)(nb * 4) * Ktiles * 512 + lane * 16;

  v8f acc[2][4] = {};
  for (int kt = 0; kt < Ktiles; ++kt) {
    v16bf a0 = ldfrag(a0p + kt * 32);
    v16bf a1 = ldfrag(a1p + kt * 32);
    if (kt + 16 < Ktiles) {
      __builtin_prefetch(a0p + (kt + 16) * 32, 0, 1);
      __builtin_prefetch(bp + (size_t)(kt + 16) * 512, 0, 1);
    }
#pragma unroll
    for (int nt = 0; nt < 4; ++nt) {
      v16bf b = ldfrag(bp + ((size_t)nt * Ktiles + kt) * 512);
      acc[0][nt] = wmma_bf16(a0, b, acc[0][nt]);
      acc[1][nt] = wmma_bf16(a1, b, acc[1][nt]);
    }
  }
#pragma unroll
  for (int mt = 0; mt < 2; ++mt)
#pragma unroll
    for (int r = 0; r < 8; ++r) {
      int row = mb * 32 + mt * 16 + r + g16 * 8;
#pragma unroll
      for (int nt = 0; nt < 4; ++nt)
        Out[(size_t)row * HID + nb * 64 + nt * 16 + l16] = acc[mt][nt][r];
    }
}

// ---------------------------------------------------------------------------
// Host launcher
// ---------------------------------------------------------------------------
extern "C" void kernel_launch(void* const* d_in, const int* in_sizes, int n_in,
                              void* d_out, int out_size, void* d_ws, size_t ws_size,
                              hipStream_t stream) {
  (void)in_sizes; (void)n_in; (void)out_size; (void)ws_size;

  const float* X  = (const float*)d_in[0];
  // d_in[1] = attention_mask: causal, reconstructed analytically in-kernel
  const float* wq = (const float*)d_in[2];
  const float* wk = (const float*)d_in[3];
  const float* wv = (const float*)d_in[4];
  const float* wo = (const float*)d_in[5];
  float* out = (float*)d_out;

  char* ws = (char*)d_ws;
  size_t off = 0;
  auto take = [&](size_t bytes) -> void* {
    void* p = ws + off;
    off += (bytes + 255) & ~(size_t)255;
    return p;
  };
  unsigned short* Xb  = (unsigned short*)take((size_t)S_LEN * HID * 2);          // 16 MB
  unsigned short* Qr  = (unsigned short*)take((size_t)NQH  * S_LEN * HD * 2);    // 16 MB
  unsigned short* Kr  = (unsigned short*)take((size_t)NKVH * S_LEN * HD * 2);    //  4 MB
  unsigned short* Vt  = (unsigned short*)take((size_t)NKVH * HD * S_LEN * 2);    //  4 MB
  unsigned short* AO  = (unsigned short*)take((size_t)S_LEN * HID * 2);          // 16 MB
  unsigned short* WqP = (unsigned short*)take((size_t)HID * HID * 2);            // 32 MB
  unsigned short* WkP = (unsigned short*)take((size_t)HID * (NKVH * HD) * 2);    //  8 MB
  unsigned short* WvP = (unsigned short*)take((size_t)HID * (NKVH * HD) * 2);    //  8 MB
  unsigned short* WoP = (unsigned short*)take((size_t)HID * HID * 2);            // 32 MB

  const int nX = S_LEN * HID;
  k_f32_to_bf16<<<(nX + 255) / 256, 256, 0, stream>>>(X, Xb, nX);

  k_pack_w<<<(HID * HID + 255) / 256, 256, 0, stream>>>(wq, WqP, HID, HID);
  k_pack_w<<<(HID * NKVH * HD + 255) / 256, 256, 0, stream>>>(wk, WkP, HID, NKVH * HD);
  k_pack_w<<<(HID * NKVH * HD + 255) / 256, 256, 0, stream>>>(wv, WvP, HID, NKVH * HD);
  k_pack_w<<<(HID * HID + 255) / 256, 256, 0, stream>>>(wo, WoP, HID, HID);

  k_qkv_rope<<<dim3(S_LEN / 32, NQH + 2 * NKVH), 32, 0, stream>>>(
      Xb, WqP, WkP, WvP, Qr, Kr, Vt);

  k_attn<<<dim3(S_LEN / 32, NQH), 32, 0, stream>>>(Qr, Kr, Vt, AO);

  k_oproj<<<dim3(HID / 64, S_LEN / 32), 32, 0, stream>>>(AO, WoP, out);
}
